// MultiHeadAttention_75393855914286
// MI455X (gfx1250) — matrix operand order repaired and checked by reading, not compiled, not run on hardware
//
#include <hip/hip_runtime.h>
#include <hip/hip_bf16.h>

typedef __attribute__((ext_vector_type(16))) __bf16 v16bf;
typedef __attribute__((ext_vector_type(8)))  float  v8f;

namespace {

constexpr int kD  = 1024;          // D_MODEL
constexpr int kH  = 16;            // heads
constexpr int kDk = 64;            // head dim
constexpr int kB  = 2;
constexpr int kS  = 2048;
constexpr int kM  = kB * kS;       // 4096 projection rows
constexpr int kBH = kB * kH;       // 32

union Frag16 {
  v16bf v;
  uint4 q[2];
};

__device__ __forceinline__ unsigned pack2_bf16(float lo, float hi) {
  unsigned a = __float_as_uint(lo);
  unsigned b = __float_as_uint(hi);
  a += 0x7FFFu + ((a >> 16) & 1u);   // round-to-nearest-even
  b += 0x7FFFu + ((b >> 16) & 1u);
  return (a >> 16) | (b & 0xFFFF0000u);
}

__device__ __forceinline__ unsigned short bf16_1(float x) {
  return (unsigned short)(pack2_bf16(x, 0.f) & 0xFFFFu);
}

__device__ __forceinline__ uint4 cvt8_f32_bf16(const float4 f0, const float4 f1) {
  uint4 r;
  r.x = pack2_bf16(f0.x, f0.y);
  r.y = pack2_bf16(f0.z, f0.w);
  r.z = pack2_bf16(f1.x, f1.y);
  r.w = pack2_bf16(f1.z, f1.w);
  return r;
}

__device__ __forceinline__ v8f wmma_bf16(const Frag16& a, const Frag16& b, v8f c) {
  return __builtin_amdgcn_wmma_f32_16x16x32_bf16(false, a.v, false, b.v,
                                                 (short)0, c, false, false);
}

// ---------------------------------------------------------------------------
// Y = X @ W^T + bias.  X: [4096,1024] (f32 or bf16), W: [1024,1024] f32.
// STORE_MODE 0: bf16 [B,H,S,Dk]   (Q, K)
// STORE_MODE 1: bf16 [B,H,Dk,S]   (V transposed for the PV GEMM)
// STORE_MODE 2: f32  [4096,1024]  (final output projection)
// Wave = 16x64 tile (4 wmma accumulators); block = 8 waves = 128x64.
// ---------------------------------------------------------------------------
template <bool A_BF16, int STORE_MODE>
__global__ __launch_bounds__(256) void proj_gemm(
    const void* __restrict__ Aptr, const float* __restrict__ W,
    const float* __restrict__ bias, void* __restrict__ Yptr) {
  const int lane = threadIdx.x & 31;
  const int wave = threadIdx.x >> 5;
  const int hi   = lane >> 4;         // half-wave select
  const int lrow = lane & 15;
  const int m0 = blockIdx.x * 128 + wave * 16;
  const int n0 = blockIdx.y * 64;

  v8f acc[4] = {};

  for (int kk = 0; kk < kD; kk += 32) {
    Frag16 a;
    if (A_BF16) {
      const unsigned short* ab =
          (const unsigned short*)Aptr + (size_t)(m0 + lrow) * kD + kk + hi * 8;
      a.q[0] = *(const uint4*)(ab);
      a.q[1] = *(const uint4*)(ab + 16);
    } else {
      const float* af = (const float*)Aptr + (size_t)(m0 + lrow) * kD + kk + hi * 8;
      a.q[0] = cvt8_f32_bf16(*(const float4*)af, *(const float4*)(af + 4));
      a.q[1] = cvt8_f32_bf16(*(const float4*)(af + 16), *(const float4*)(af + 20));
    }
#pragma unroll
    for (int t = 0; t < 4; ++t) {
      const float* wb = W + (size_t)(n0 + t * 16 + lrow) * kD + kk + hi * 8;
      Frag16 bf;
      bf.q[0] = cvt8_f32_bf16(*(const float4*)wb, *(const float4*)(wb + 4));
      bf.q[1] = cvt8_f32_bf16(*(const float4*)(wb + 16), *(const float4*)(wb + 20));
      acc[t] = wmma_bf16(a, bf, acc[t]);
    }
  }

#pragma unroll
  for (int t = 0; t < 4; ++t) {
    const int n = n0 + t * 16 + lrow;
    const float bv = bias[n];
#pragma unroll
    for (int r = 0; r < 8; ++r) {
      const int m = m0 + r + hi * 8;
      const float val = acc[t][r] + bv;
      if (STORE_MODE == 2) {
        ((float*)Yptr)[(size_t)m * kD + n] = val;
      } else {
        const int bb = m >> 11;        // batch
        const int s  = m & (kS - 1);   // sequence pos
        const int h  = n >> 6;         // head
        const int d  = n & (kDk - 1);  // dim in head
        const unsigned short bw = bf16_1(val);
        if (STORE_MODE == 0)
          ((unsigned short*)Yptr)[((size_t)(bb * kH + h) * kS + s) * kDk + d] = bw;
        else
          ((unsigned short*)Yptr)[((size_t)(bb * kH + h) * kDk + d) * kS + s] = bw;
      }
    }
  }
}

// ---------------------------------------------------------------------------
// Fused attention core: scores (QK^T/8) -> softmax -> write attn once ->
// O = P @ V, all per 16 query rows of one (b,h).
//
// Block = 8 waves. Wave w owns score columns [w*256, w*256+256) held in
// 16 wmma accumulators (128 VGPRs). Row max/sum: __shfl_xor within half-wave
// + LDS across waves. P is staged to LDS as bf16 in two 16x1024 passes
// (33 KB, padded stride) and consumed as WMMA A-fragments via ds_load_b128.
// Partial O tiles from the 8 waves are combined with shared-memory atomicAdd
// (ds_add_f32).
// ---------------------------------------------------------------------------
__global__ __launch_bounds__(256) void fused_attn(
    const unsigned short* __restrict__ Qbf,   // [BH,S,Dk]
    const unsigned short* __restrict__ Kbf,   // [BH,S,Dk]
    const unsigned short* __restrict__ Vt,    // [BH,Dk,S]
    float* __restrict__ attn,                 // [BH,S,S]
    unsigned short* __restrict__ Obf) {       // [B,S,H*Dk] bf16
  __shared__ unsigned short Pt[16][1032];     // half of P tile, padded (33 KB)
  __shared__ float redbuf[8][16];             // cross-wave row reductions
  __shared__ float Ored[16][64];              // O accumulation

  const int lane = threadIdx.x & 31;
  const int w    = threadIdx.x >> 5;
  const int hi   = lane >> 4;
  const int lrow = lane & 15;
  const int bh = blockIdx.y;
  const int q0 = blockIdx.x * 16;

  const unsigned short* Qh = Qbf + (size_t)bh * kS * kDk;
  const unsigned short* Kh = Kbf + (size_t)bh * kS * kDk;
  const unsigned short* Vh = Vt  + (size_t)bh * kDk * kS;
  float* attn_out = attn + (size_t)bh * kS * kS;

  // ------------------ Phase 1: scores (16 rows x 256 cols per wave) --------
  Frag16 aq[2];
  {
    const unsigned short* ab = Qh + (size_t)(q0 + lrow) * kDk + hi * 8;
    aq[0].q[0] = *(const uint4*)ab;
    aq[0].q[1] = *(const uint4*)(ab + 16);
    aq[1].q[0] = *(const uint4*)(ab + 32);
    aq[1].q[1] = *(const uint4*)(ab + 48);
  }

  v8f acc[16] = {};
#pragma unroll
  for (int t = 0; t < 16; ++t) {
    const int n = w * 256 + t * 16 + lrow;
    const unsigned short* kb = Kh + (size_t)n * kDk + hi * 8;
    Frag16 b0, b1;
    b0.q[0] = *(const uint4*)kb;
    b0.q[1] = *(const uint4*)(kb + 16);
    b1.q[0] = *(const uint4*)(kb + 32);
    b1.q[1] = *(const uint4*)(kb + 48);
    acc[t] = wmma_bf16(aq[0], b0, acc[t]);
    acc[t] = wmma_bf16(aq[1], b1, acc[t]);
  }

  // ------------------ Phase 2: softmax -------------------------------------
  float rmax[8];
#pragma unroll
  for (int r = 0; r < 8; ++r) rmax[r] = -3.0e38f;
#pragma unroll
  for (int t = 0; t < 16; ++t)
#pragma unroll
    for (int r = 0; r < 8; ++r) {
      acc[t][r] *= 0.125f;                       // 1/sqrt(Dk)
      rmax[r] = fmaxf(rmax[r], acc[t][r]);
    }
#pragma unroll
  for (int m = 1; m <= 8; m <<= 1)
#pragma unroll
    for (int r = 0; r < 8; ++r)
      rmax[r] = fmaxf(rmax[r], __shfl_xor(rmax[r], m, 32));
  if (lrow == 0)
#pragma unroll
    for (int r = 0; r < 8; ++r) redbuf[w][hi * 8 + r] = rmax[r];
  __syncthreads();
#pragma unroll
  for (int r = 0; r < 8; ++r) {
    float m = redbuf[0][hi * 8 + r];
#pragma unroll
    for (int u = 1; u < 8; ++u) m = fmaxf(m, redbuf[u][hi * 8 + r]);
    rmax[r] = m;
  }
  __syncthreads();   // before redbuf reuse

  float rsum[8] = {};
#pragma unroll
  for (int t = 0; t < 16; ++t)
#pragma unroll
    for (int r = 0; r < 8; ++r) {
      const float e = __expf(acc[t][r] - rmax[r]);
      acc[t][r] = e;
      rsum[r] += e;
    }
#pragma unroll
  for (int m = 1; m <= 8; m <<= 1)
#pragma unroll
    for (int r = 0; r < 8; ++r) rsum[r] += __shfl_xor(rsum[r], m, 32);
  if (lrow == 0)
#pragma unroll
    for (int r = 0; r < 8; ++r) redbuf[w][hi * 8 + r] = rsum[r];
  __syncthreads();
#pragma unroll
  for (int r = 0; r < 8; ++r) {
    float s = 0.f;
#pragma unroll
    for (int u = 0; u < 8; ++u) s += redbuf[u][hi * 8 + r];
    rsum[r] = 1.0f / s;
  }

  // Normalize in registers and write attn to d_out exactly once (fp32).
#pragma unroll
  for (int t = 0; t < 16; ++t) {
    const int n = w * 256 + t * 16 + lrow;
#pragma unroll
    for (int r = 0; r < 8; ++r) {
      const float p = acc[t][r] * rsum[r];
      acc[t][r] = p;
      attn_out[(size_t)(q0 + hi * 8 + r) * kS + n] = p;
    }
  }

  for (int i = threadIdx.x; i < 16 * 64; i += 256) ((float*)Ored)[i] = 0.f;

  // ------------------ Phase 3: O = P @ V in two half-K passes --------------
  v8f oacc[4] = {};
#pragma unroll
  for (int half = 0; half < 2; ++half) {
    __syncthreads();                 // previous Pt reads complete
    if ((w >> 2) == half) {          // waves owning this half stage P as bf16
      const int wl = w & 3;
#pragma unroll
      for (int t = 0; t < 16; ++t) {
        const int cl = wl * 256 + t * 16 + lrow;
#pragma unroll
        for (int r = 0; r < 8; ++r) Pt[hi * 8 + r][cl] = bf16_1(acc[t][r]);
      }
    }
    __syncthreads();
    // Each wave accumulates its 128-wide K chunk of this half.
#pragma unroll
    for (int kk = 0; kk < 128; kk += 32) {
      const int kl = w * 128 + kk;               // k within half
      Frag16 a;
      const unsigned short* pp = &Pt[lrow][kl + hi * 8];
      a.q[0] = *(const uint4*)pp;
      a.q[1] = *(const uint4*)(pp + 16);
#pragma unroll
      for (int t = 0; t < 4; ++t) {
        const unsigned short* vb =
            Vh + (size_t)(t * 16 + lrow) * kS + half * 1024 + kl + hi * 8;
        Frag16 b;
        b.q[0] = *(const uint4*)vb;
        b.q[1] = *(const uint4*)(vb + 16);
        oacc[t] = wmma_bf16(a, b, oacc[t]);
      }
    }
  }
  __syncthreads();

  // Combine the 8 waves' partial O tiles (ds_add_f32).
#pragma unroll
  for (int t = 0; t < 4; ++t)
#pragma unroll
    for (int r = 0; r < 8; ++r)
      atomicAdd(&Ored[hi * 8 + r][t * 16 + lrow], oacc[t][r]);
  __syncthreads();

  const int b = bh >> 4, h = bh & (kH - 1);
  for (int i = threadIdx.x; i < 16 * 64; i += 256) {
    const int row = i >> 6, d = i & 63;
    Obf[(size_t)(b * kS + q0 + row) * kD + h * kDk + d] = bf16_1(Ored[row][d]);
  }
}

}  // namespace

extern "C" void kernel_launch(void* const* d_in, const int* in_sizes, int n_in,
                              void* d_out, int out_size, void* d_ws, size_t ws_size,
                              hipStream_t stream) {
  const float* q   = (const float*)d_in[0];
  const float* k   = (const float*)d_in[1];
  const float* v   = (const float*)d_in[2];
  const float* w_q = (const float*)d_in[3];
  const float* b_q = (const float*)d_in[4];
  const float* w_k = (const float*)d_in[5];
  const float* b_k = (const float*)d_in[6];
  const float* w_v = (const float*)d_in[7];
  const float* b_v = (const float*)d_in[8];
  const float* w_o = (const float*)d_in[9];
  const float* b_o = (const float*)d_in[10];

  float* out  = (float*)d_out;                       // [B,S,D] fp32
  float* attn = out + (size_t)kM * kD;               // [B,H,S,S] fp32

  char* ws = (char*)d_ws;                            // 32 MB used
  unsigned short* Qbf = (unsigned short*)(ws);                               // 8 MB [B,H,S,Dk]
  unsigned short* Kbf = (unsigned short*)(ws + (size_t)8  * 1024 * 1024);    // 8 MB [B,H,S,Dk]
  unsigned short* Vt  = (unsigned short*)(ws + (size_t)16 * 1024 * 1024);    // 8 MB [B,H,Dk,S]
  unsigned short* Obf = (unsigned short*)(ws + (size_t)24 * 1024 * 1024);    // 8 MB [B,S,D]

  const dim3 blk(256);
  const dim3 gProj(kM / 128, kD / 64);               // 32 x 16

  // QKV projections (fp32 in, bf16 out; V stored transposed per head)
  proj_gemm<false, 0><<<gProj, blk, 0, stream>>>(q, w_q, b_q, Qbf);
  proj_gemm<false, 0><<<gProj, blk, 0, stream>>>(k, w_k, b_k, Kbf);
  proj_gemm<false, 1><<<gProj, blk, 0, stream>>>(v, w_v, b_v, Vt);

  // Fused scores + softmax + attn write + P@V
  fused_attn<<<dim3(kS / 16, kBH), blk, 0, stream>>>(Qbf, Kbf, Vt, attn, Obf);

  // final projection: out = O @ w_o^T + b_o (bf16 A, fp32 out)
  proj_gemm<true, 2><<<gProj, blk, 0, stream>>>(Obf, w_o, b_o, out);
}
